// _CategoryDense_29978871726697
// MI455X (gfx1250) — matrix operand order repaired and checked by reading, not compiled, not run on hardware
//
#include <hip/hip_runtime.h>

// CategoryDense: out[b,c,o] = sum_i inp[b,c,i] * wgt[c,i,o] + bias[c,o]
// B=256, C=919, IN=400, OUT=100, fp32 in/out.
// bf16x3 split-precision WMMA (v_wmma_f32_16x16x32_bf16).
// A tile staged in LDS (hi/lo, zero-padded K) and shared by 7 waves;
// B loaded unconditionally (clamped addresses; A zero-padding nullifies tail,
// unstored D columns absorb n>=100 garbage) -> no exec-mask branching.

#define B_DIM   256
#define C_DIM   919
#define IN_DIM  400
#define OUT_DIM 100
#define KCHUNKS 13    // ceil(400/32), tail zero-padded on the A side
#define KSTRIDE 424   // LDS row stride (bf16): 212 dwords % 64 banks = 20 -> min-conflict b128
#define THREADS 224   // 7 wave32 waves, one per 16-wide N tile

typedef __bf16 v8bf  __attribute__((ext_vector_type(8)));
typedef __bf16 v16bf __attribute__((ext_vector_type(16)));
typedef float  v8f   __attribute__((ext_vector_type(8)));

__global__ __launch_bounds__(THREADS)
void _CategoryDense_wmma_bf16x3(const float* __restrict__ inp,
                                const float* __restrict__ wgt,
                                const float* __restrict__ bias,
                                float* __restrict__ out) {
    __shared__ __bf16 Ah[16 * KSTRIDE];
    __shared__ __bf16 Al[16 * KSTRIDE];

    const int c  = blockIdx.x;        // category
    const int b0 = blockIdx.y << 4;   // M tile base (16 batch rows)

    // ---- stage A tile: fp32 -> bf16 hi/lo in LDS, K zero-padded (branch-free) ----
    for (int row = 0; row < 16; ++row) {
        const float* src = inp + ((long)(b0 + row) * C_DIM + c) * IN_DIM;
        for (int k = threadIdx.x; k < KSTRIDE; k += THREADS) {
            const int  kc = (k < IN_DIM) ? k : (IN_DIM - 1);   // safe address
            float v = src[kc];
            v = (k < IN_DIM) ? v : 0.0f;                       // zero pad (v_cndmask)
            __bf16 hi = (__bf16)v;
            Ah[row * KSTRIDE + k] = hi;
            Al[row * KSTRIDE + k] = (__bf16)(v - (float)hi);
        }
    }
    __syncthreads();

    const int wave = threadIdx.x >> 5;   // 0..6 -> N tile
    const int lane = threadIdx.x & 31;
    const int half = lane >> 4;
    const int lmod = lane & 15;
    const int n    = (wave << 4) + lmod;          // D/B column this lane owns
    const bool nvalid = (n < OUT_DIM);
    const int  nc = nvalid ? n : (OUT_DIM - 1);   // clamped safe column

    const float* Wn = wgt + (long)c * IN_DIM * OUT_DIM + nc;
    const int rowoff = lmod * KSTRIDE;

    v8f acc = {};

    #pragma unroll 4
    for (int chunk = 0; chunk < KCHUNKS; ++chunk) {
        const int kb = chunk << 5;

        // ---- A fragments (contiguous K layout): lane M = lmod,
        //      runs [kb+16*half .. +7] and [kb+16*half+8 .. +7] ----
        const int r1 = kb + (half << 4);
        v8bf ah1 = *(const v8bf*)(&Ah[rowoff + r1]);
        v8bf ah2 = *(const v8bf*)(&Ah[rowoff + r1 + 8]);
        v8bf al1 = *(const v8bf*)(&Al[rowoff + r1]);
        v8bf al2 = *(const v8bf*)(&Al[rowoff + r1 + 8]);
        v16bf ah, al;
        #pragma unroll
        for (int i = 0; i < 8; ++i) {
            ah[i] = ah1[i]; ah[i + 8] = ah2[i];
            al[i] = al1[i]; al[i + 8] = al2[i];
        }

        // ---- B fragments: lane N = n, K = kcol..kcol+15. Unconditional loads,
        //      address-clamped only; 64B coalesced across lanes 0..15 / 16..31 ----
        int kcol = kb + (half << 4);
        kcol = (kcol > IN_DIM - 16) ? (IN_DIM - 16) : kcol;   // tail clamp (A zeros nullify)
        const float* p = Wn + (long)kcol * OUT_DIM;

        v16bf bh, bl;
        #pragma unroll
        for (int e = 0; e < 16; ++e) {
            const float w = p[(long)e * OUT_DIM];
            __bf16 h = (__bf16)w;
            bh[e] = h;
            bl[e] = (__bf16)(w - (float)h);
        }

        // ---- bf16x3 fp32 emulation: acc += al*bh + ah*bl + ah*bh ----
        acc = __builtin_amdgcn_wmma_f32_16x16x32_bf16(false, al, false, bh,
                                                      (short)0, acc, false, false);
        acc = __builtin_amdgcn_wmma_f32_16x16x32_bf16(false, ah, false, bl,
                                                      (short)0, acc, false, false);
        acc = __builtin_amdgcn_wmma_f32_16x16x32_bf16(false, ah, false, bh,
                                                      (short)0, acc, false, false);
    }

    // ---- epilogue: D layout (M = r + 8*half, N = n) + bias ----
    if (nvalid) {
        const float bv = bias[(long)c * OUT_DIM + n];
        #pragma unroll
        for (int r = 0; r < 8; ++r) {
            const int m = b0 + r + (half << 3);
            out[((long)m * C_DIM + c) * OUT_DIM + n] = acc[r] + bv;
        }
    }
}

extern "C" void kernel_launch(void* const* d_in, const int* in_sizes, int n_in,
                              void* d_out, int out_size, void* d_ws, size_t ws_size,
                              hipStream_t stream) {
    const float* inp  = (const float*)d_in[0];   // (B, C, IN)  fp32
    const float* wgt  = (const float*)d_in[1];   // (1, C, IN, OUT) fp32
    const float* bias = (const float*)d_in[2];   // (1, C, OUT) fp32
    float* out = (float*)d_out;                  // (B, C, OUT) fp32

    dim3 grid(C_DIM, B_DIM / 16);                // 919 categories x 16 M-tiles
    _CategoryDense_wmma_bf16x3<<<grid, THREADS, 0, stream>>>(inp, wgt, bias, out);
}
